// MemristorCNN_88046829568547
// MI455X (gfx1250) — matrix operand order repaired and checked by reading, not compiled, not run on hardware
//
#include <hip/hip_runtime.h>

typedef __attribute__((ext_vector_type(16))) _Float16 v16h;
typedef __attribute__((ext_vector_type(8)))  float    v8f;
typedef __attribute__((ext_vector_type(4)))  int      v4i;
typedef __attribute__((ext_vector_type(4)))  float    v4f;

#define IMG   224
#define NB    32          // batch
#define C1    16
#define C2    32
#define H1S   112         // after pool1
#define H2S   56          // after pool2
#define K_FEAT 100352     // 32*56*56
#define NOUT1 512
#define NVALS 4096
#define KCHUNKS 14        // blockIdx.y
#define WAVES_PER_BLK 8
#define STEPS 28          // 14 * 8 * 28 * 32 == 100352

// ---------------- conv1 + relu + 2x2 maxpool ----------------
__global__ void conv1_pool(const float* __restrict__ x,
                           const float* __restrict__ w,
                           const float* __restrict__ bias,
                           float* __restrict__ h1) {
  int idx = blockIdx.x * blockDim.x + threadIdx.x;
  const int total = NB * C1 * H1S * H1S;
  if (idx >= total) return;
  int px = idx % H1S;
  int py = (idx / H1S) % H1S;
  int c  = (idx / (H1S * H1S)) % C1;
  int b  = idx / (H1S * H1S * C1);

  float wv[9];
#pragma unroll
  for (int i = 0; i < 9; ++i) wv[i] = w[c * 9 + i];

  const float* xb = x + (size_t)b * IMG * IMG;
  int r0 = 2 * py - 1, q0 = 2 * px - 1;
  float patch[4][4];
#pragma unroll
  for (int r = 0; r < 4; ++r) {
    int rr = r0 + r;
    bool rok = (rr >= 0) && (rr < IMG);
#pragma unroll
    for (int cc = 0; cc < 4; ++cc) {
      int qq = q0 + cc;
      patch[r][cc] = (rok && qq >= 0 && qq < IMG) ? xb[rr * IMG + qq] : 0.f;
    }
  }
  float s00 = 0.f, s01 = 0.f, s10 = 0.f, s11 = 0.f;
#pragma unroll
  for (int dy = 0; dy < 3; ++dy)
#pragma unroll
    for (int dx = 0; dx < 3; ++dx) {
      float ww = wv[dy * 3 + dx];
      s00 = fmaf(patch[dy    ][dx    ], ww, s00);
      s01 = fmaf(patch[dy    ][dx + 1], ww, s01);
      s10 = fmaf(patch[dy + 1][dx    ], ww, s10);
      s11 = fmaf(patch[dy + 1][dx + 1], ww, s11);
    }
  // relu and max commute (both monotone): pool(relu(conv+b)) == relu(max(conv)+b)
  float m = fmaxf(fmaxf(s00, s01), fmaxf(s10, s11)) + bias[c];
  h1[idx] = m > 0.f ? m : 0.f;
}

// ---------------- conv2 + relu + 2x2 maxpool ----------------
__global__ void conv2_pool(const float* __restrict__ h1,
                           const float* __restrict__ w,
                           const float* __restrict__ bias,
                           float* __restrict__ h2) {
  int idx = blockIdx.x * blockDim.x + threadIdx.x;
  const int total = NB * C2 * H2S * H2S;
  if (idx >= total) return;
  int px = idx % H2S;
  int py = (idx / H2S) % H2S;
  int c  = (idx / (H2S * H2S)) % C2;
  int b  = idx / (H2S * H2S * C2);

  int r0 = 2 * py - 1, q0 = 2 * px - 1;
  const float* wc = w + c * C1 * 9;
  float s00 = 0.f, s01 = 0.f, s10 = 0.f, s11 = 0.f;

  for (int ic = 0; ic < C1; ++ic) {
    const float* plane = h1 + ((size_t)(b * C1 + ic)) * H1S * H1S;
    float wv[9];
#pragma unroll
    for (int i = 0; i < 9; ++i) wv[i] = wc[ic * 9 + i];
    float patch[4][4];
#pragma unroll
    for (int r = 0; r < 4; ++r) {
      int rr = r0 + r;
      bool rok = (rr >= 0) && (rr < H1S);
#pragma unroll
      for (int cc = 0; cc < 4; ++cc) {
        int qq = q0 + cc;
        patch[r][cc] = (rok && qq >= 0 && qq < H1S) ? plane[rr * H1S + qq] : 0.f;
      }
    }
#pragma unroll
    for (int dy = 0; dy < 3; ++dy)
#pragma unroll
      for (int dx = 0; dx < 3; ++dx) {
        float ww = wv[dy * 3 + dx];
        s00 = fmaf(patch[dy    ][dx    ], ww, s00);
        s01 = fmaf(patch[dy    ][dx + 1], ww, s01);
        s10 = fmaf(patch[dy + 1][dx    ], ww, s10);
        s11 = fmaf(patch[dy + 1][dx + 1], ww, s11);
      }
  }
  float m = fmaxf(fmaxf(s00, s01), fmaxf(s10, s11)) + bias[c];
  h2[idx] = m > 0.f ? m : 0.f;  // flattened layout [b][c][y][x] == reference reshape order
}

// A-tile loader: 16x32 f16 A-layout. Lane (lo half, M=lane): K {khalf..khalf+7},
// {khalf+16..khalf+23} packed as elements 0..7 / 8..15.
__device__ __forceinline__ v16h load_a_tile(const float* ap) {
  v4f p0 = ((const v4f*)ap)[0];
  v4f p1 = ((const v4f*)ap)[1];
  v4f q0 = ((const v4f*)(ap + 16))[0];
  v4f q1 = ((const v4f*)(ap + 16))[1];
  v16h a;
  a[0]  = (_Float16)p0[0]; a[1]  = (_Float16)p0[1]; a[2]  = (_Float16)p0[2]; a[3]  = (_Float16)p0[3];
  a[4]  = (_Float16)p1[0]; a[5]  = (_Float16)p1[1]; a[6]  = (_Float16)p1[2]; a[7]  = (_Float16)p1[3];
  a[8]  = (_Float16)q0[0]; a[9]  = (_Float16)q0[1]; a[10] = (_Float16)q0[2]; a[11] = (_Float16)q0[3];
  a[12] = (_Float16)q1[0]; a[13] = (_Float16)q1[1]; a[14] = (_Float16)q1[2]; a[15] = (_Float16)q1[3];
  return a;
}

// ---------------- fc1: codebook-gather GEMM on WMMA ----------------
// grid: (out_tile=32, kchunk=14), block = 256 (8 waves).
// Each wave holds TWO accumulators (batch rows 0-15 and 16-31) so the
// 205 MB w_idx1 stream is fetched from HBM exactly once (non-temporally,
// keeping L2 for the h2 activations that are re-read per output tile).
__global__ void __launch_bounds__(256)
fc1_wmma(const float* __restrict__ h2,
         const int*   __restrict__ w_idx1,
         const float* __restrict__ values,
         float*       __restrict__ fc1_part) {
  __shared__ _Float16 sval[NVALS];          // 8 KB f16 codebook in LDS
  for (int i = threadIdx.x; i < NVALS; i += 256)
    sval[i] = (_Float16)values[i];
  __syncthreads();

  const int lane = threadIdx.x & 31;
  const int wave = threadIdx.x >> 5;
  const int ot   = blockIdx.x;              // output tile: neurons [ot*16, ot*16+16)
  const int kb   = blockIdx.y;              // K chunk

  const int kbase0 = (kb * WAVES_PER_BLK + wave) * (STEPS * 32);

  const int m     = lane & 15;
  const int khalf = (lane >> 4) * 8;        // A K-half select
  const int kb2   = (lane >> 4) * 8;        // B K-half select

  const float* arow0 = h2     + (size_t)m        * K_FEAT;   // batch rows 0-15
  const float* arow1 = h2     + (size_t)(16 + m) * K_FEAT;   // batch rows 16-31
  const int*   irow  = w_idx1 + (size_t)(ot * 16 + m) * K_FEAT;

  v8f acc0 = {};
  v8f acc1 = {};
  for (int s = 0; s < STEPS; ++s) {
    const int kbase = kbase0 + s * 32;

    // ---- B tile: stream 16 indices (64B) per lane NT, gather f16 codebook from LDS ----
    const int* ip = irow + kbase + kb2;
    __builtin_prefetch(ip + 32, 0, 0);                 // next step's line -> global_prefetch_b8
    v4i i0 = __builtin_nontemporal_load((const v4i*)ip);
    v4i i1 = __builtin_nontemporal_load(((const v4i*)ip) + 1);
    v4i i2 = __builtin_nontemporal_load(((const v4i*)ip) + 4);
    v4i i3 = __builtin_nontemporal_load(((const v4i*)ip) + 5);
    v16h bv;
    bv[0]  = sval[i0[0]]; bv[1]  = sval[i0[1]]; bv[2]  = sval[i0[2]]; bv[3]  = sval[i0[3]];
    bv[4]  = sval[i1[0]]; bv[5]  = sval[i1[1]]; bv[6]  = sval[i1[2]]; bv[7]  = sval[i1[3]];
    bv[8]  = sval[i2[0]]; bv[9]  = sval[i2[1]]; bv[10] = sval[i2[2]]; bv[11] = sval[i2[3]];
    bv[12] = sval[i3[0]]; bv[13] = sval[i3[1]]; bv[14] = sval[i3[2]]; bv[15] = sval[i3[3]];

    // ---- A tiles for both batch halves (L2-resident activations) ----
    v16h a0 = load_a_tile(arow0 + kbase + khalf);
    v16h a1 = load_a_tile(arow1 + kbase + khalf);

    acc0 = __builtin_amdgcn_wmma_f32_16x16x32_f16(
        false, a0, false, bv, (short)0, acc0, false, false);
    acc1 = __builtin_amdgcn_wmma_f32_16x16x32_f16(
        false, a1, false, bv, (short)0, acc1, false, false);
  }

  // D layout: lane 0-15 -> N=lane, M=v; lane 16-31 -> N=lane-16, M=v+8
  const int wchunk = kb * WAVES_PER_BLK + wave;  // 0..111
  const int n = lane & 15;
#pragma unroll
  for (int v = 0; v < 8; ++v) {
    int mrow = (lane < 16) ? v : (v + 8);
    fc1_part[((size_t)wchunk * 32 + mrow)      * NOUT1 + ot * 16 + n] = acc0[v];
    fc1_part[((size_t)wchunk * 32 + 16 + mrow) * NOUT1 + ot * 16 + n] = acc1[v];
  }
}

// ---------------- deterministic fixed-order reduction of 112 partials + bias ----------------
__global__ void fc1_reduce(const float* __restrict__ fc1_part,
                           const float* __restrict__ b1,
                           float* __restrict__ fc1_out) {
  int t = blockIdx.x * blockDim.x + threadIdx.x;   // 0 .. 32*512-1
  if (t >= NB * NOUT1) return;
  float s = b1[t & (NOUT1 - 1)];
  for (int w = 0; w < KCHUNKS * WAVES_PER_BLK; ++w)
    s += fc1_part[(size_t)w * NB * NOUT1 + t];
  fc1_out[t] = s;                                  // relu deferred to fc2
}

// ---------------- fc2 (tiny) ----------------
__global__ void fc2_kernel(const float* __restrict__ fc1,
                           const int*   __restrict__ w_idx2,
                           const float* __restrict__ values,
                           const float* __restrict__ b2,
                           float* __restrict__ out) {
  int t = threadIdx.x;
  if (t >= NB * 4) return;
  int b = t >> 2, o = t & 3;
  float s = b2[o];
  const int*   irow = w_idx2 + o * NOUT1;
  const float* hrow = fc1 + b * NOUT1;
  for (int j = 0; j < NOUT1; ++j) {
    float h = hrow[j];
    h = h > 0.f ? h : 0.f;
    s = fmaf(h, values[irow[j]], s);
  }
  out[b * 4 + o] = s;
}

extern "C" void kernel_launch(void* const* d_in, const int* in_sizes, int n_in,
                              void* d_out, int out_size, void* d_ws, size_t ws_size,
                              hipStream_t stream) {
  const float* x      = (const float*)d_in[0];
  const float* c1w    = (const float*)d_in[1];
  const float* c1b    = (const float*)d_in[2];
  const float* c2w    = (const float*)d_in[3];
  const float* c2b    = (const float*)d_in[4];
  const float* values = (const float*)d_in[5];
  const int*   w_idx1 = (const int*)  d_in[6];
  const float* fc1_b  = (const float*)d_in[7];
  const int*   w_idx2 = (const int*)  d_in[8];
  const float* fc2_b  = (const float*)d_in[9];
  float* out = (float*)d_out;

  // workspace layout (f32):
  //   h1:       32*16*112*112 = 6,422,528   (25.7 MB)
  //   h2:       32*100352     = 3,211,264   (12.8 MB)
  //   fc1_part: 112*32*512    = 1,835,008   ( 7.3 MB)
  //   fc1_out:  32*512        =    16,384   (64 KB)
  float* h1       = (float*)d_ws;
  float* h2       = h1 + (size_t)NB * C1 * H1S * H1S;
  float* fc1_part = h2 + (size_t)NB * K_FEAT;
  float* fc1_out  = fc1_part + (size_t)KCHUNKS * WAVES_PER_BLK * NB * NOUT1;

  int n1 = NB * C1 * H1S * H1S;
  conv1_pool<<<(n1 + 255) / 256, 256, 0, stream>>>(x, c1w, c1b, h1);

  int n2 = NB * C2 * H2S * H2S;
  conv2_pool<<<(n2 + 255) / 256, 256, 0, stream>>>(h1, c2w, c2b, h2);

  dim3 g(32, KCHUNKS);      // out tiles x K chunks; both batch tiles per wave
  fc1_wmma<<<g, 256, 0, stream>>>(h2, w_idx1, values, fc1_part);

  int nr = NB * NOUT1;
  fc1_reduce<<<(nr + 255) / 256, 256, 0, stream>>>(fc1_part, fc1_b, fc1_out);

  fc2_kernel<<<1, 128, 0, stream>>>(fc1_out, w_idx2, values, fc2_b, out);
}
